// RQVAE_39900246180071
// MI455X (gfx1250) — matrix operand order repaired and checked by reading, not compiled, not run on hardware
//
#include <hip/hip_runtime.h>
#include <hip/hip_bf16.h>
#include <cstdint>

// Problem constants (match reference)
#define NQ 4
#define KC 1024
#define DD 512
#define NN 32768

typedef __attribute__((ext_vector_type(16))) __bf16 v16bf;
typedef __attribute__((ext_vector_type(8)))  __bf16 v8bf;
typedef __attribute__((ext_vector_type(4)))  __bf16 v4bf;
typedef __attribute__((ext_vector_type(2)))  __bf16 v2bf;
typedef __attribute__((ext_vector_type(8)))  float  v8f;

union ABfrag { v16bf v; v8bf h[2]; };

// ---------------------------------------------------------------------------
// Codebook prep: e2[l*K+k] = ||E_lk||^2 ; EbfT[l][d][k] = bf16(E[l][k][d])
// (transposed so each k row of a WMMA B-fragment is 32 contiguous bytes across 16 lanes)
// ---------------------------------------------------------------------------
__global__ __launch_bounds__(256) void prep_codebooks(const float* __restrict__ cb,
    __bf16* __restrict__ EbfT, float* __restrict__ e2) {
  __shared__ float red[256];
  int bid = blockIdx.x;              // l*KC + k
  int l = bid >> 10, k = bid & (KC - 1);
  const float* Erow = cb + (size_t)bid * DD;
  __bf16* T = EbfT + (size_t)l * DD * KC;
  float sq = 0.f;
  for (int d = threadIdx.x; d < DD; d += 256) {
    float v = Erow[d];
    T[(size_t)d * KC + k] = (__bf16)v;
    sq += v * v;
  }
  red[threadIdx.x] = sq; __syncthreads();
  for (int s = 128; s > 0; s >>= 1) {
    if (threadIdx.x < s) red[threadIdx.x] += red[threadIdx.x + s];
    __syncthreads();
  }
  if (threadIdx.x == 0) e2[bid] = red[0];
}

// ---------------------------------------------------------------------------
// Init: residual = x, xhat = 0, rbf = bf16(x), rn2[n] = ||x_n||^2
// ---------------------------------------------------------------------------
__global__ __launch_bounds__(256) void init_x(const float* __restrict__ x,
    float* __restrict__ resid, float* __restrict__ xhat,
    __bf16* __restrict__ rbf, float* __restrict__ rn2) {
  __shared__ float red[256];
  int n = blockIdx.x, tid = threadIdx.x;
  float sq = 0.f;
  for (int d = tid; d < DD; d += 256) {
    size_t o = (size_t)n * DD + d;
    float v = x[o];
    resid[o] = v;
    xhat[o]  = 0.f;
    rbf[o]   = (__bf16)v;
    sq += v * v;
  }
  red[tid] = sq; __syncthreads();
  for (int s = 128; s > 0; s >>= 1) {
    if (tid < s) red[tid] += red[tid + s];
    __syncthreads();
  }
  if (tid == 0) rn2[n] = red[0];
}

// ---------------------------------------------------------------------------
// WMMA distance GEMM + exp epilogue:
//   M[n][k] = exp(-(rn2[n] - 2*dot(r_n, E_k) + e2[k]))   (eps = 1.0)
// One wave computes a 16x64 tile: 1 A-frag reused over 4 B-frags per k-step.
// M stored as bf16: halves sinkhorn traffic and keeps M (64MB) L2-resident
// together with residual (64MB) + rbf (32MB) inside the 192MB global L2.
// ---------------------------------------------------------------------------
__global__ __launch_bounds__(256) void dist_exp_gemm(const __bf16* __restrict__ rbf,
    const __bf16* __restrict__ EbfT, const float* __restrict__ rn2,
    const float* __restrict__ e2, __bf16* __restrict__ Mmat) {
  int lane   = threadIdx.x & 31;
  int gw     = blockIdx.x * 8 + (threadIdx.x >> 5);  // 32768 waves total
  int tile_m = gw >> 4;                               // 0..2047 (row tile)
  int tile_n = (gw & 15) * 64;                        // col base (0..960)

  // A fragment addressing per ISA 16-bit A 16x32 layout:
  // lanes 0-15: rows 0-15, K chunks [0..7] & [16..23]
  // lanes 16-31: rows 0-15, K chunks [8..15] & [24..31]
  int arow  = lane & 15;
  int abase = (lane < 16) ? 0 : 8;
  const __bf16* Aptr  = rbf + (size_t)(tile_m * 16 + arow) * DD;
  const __bf16* Bbase = EbfT + tile_n;

  v8f z = {0.f,0.f,0.f,0.f,0.f,0.f,0.f,0.f};
  v8f c[4] = {z, z, z, z};

  for (int kk = 0; kk < DD; kk += 32) {
    ABfrag a;
    a.h[0] = *(const v8bf*)(Aptr + kk + abase);
    a.h[1] = *(const v8bf*)(Aptr + kk + abase + 16);
    // B: lane l holds column tile_n + (l&15), K rows in the same order as A
    const __bf16* Bcol = Bbase + (lane & 15);
    v16bf b0, b1, b2, b3;
    #pragma unroll
    for (int s = 0; s < 16; ++s) {
      size_t ko = (size_t)(kk + abase + (s & 7) + ((s >> 3) << 4)) * KC;
      b0[s] = Bcol[ko];
      b1[s] = Bcol[ko + 16];
      b2[s] = Bcol[ko + 32];
      b3[s] = Bcol[ko + 48];
    }
    c[0] = __builtin_amdgcn_wmma_f32_16x16x32_bf16(false, a.v, false, b0, (short)0, c[0], false, false);
    c[1] = __builtin_amdgcn_wmma_f32_16x16x32_bf16(false, a.v, false, b1, (short)0, c[1], false, false);
    c[2] = __builtin_amdgcn_wmma_f32_16x16x32_bf16(false, a.v, false, b2, (short)0, c[2], false, false);
    c[3] = __builtin_amdgcn_wmma_f32_16x16x32_bf16(false, a.v, false, b3, (short)0, c[3], false, false);
  }

  // C/D layout: lane<16 -> rows tile_m*16 + v, lane>=16 -> rows tile_m*16 + 8 + v
  int col0  = tile_n + (lane & 15);
  int rbase = tile_m * 16 + ((lane < 16) ? 0 : 8);
  float r2v[8];
  #pragma unroll
  for (int v = 0; v < 8; ++v) r2v[v] = rn2[rbase + v];
  #pragma unroll
  for (int t = 0; t < 4; ++t) {
    float e2v = e2[col0 + 16 * t];
    #pragma unroll
    for (int v = 0; v < 8; ++v) {
      float val = __expf(2.0f * c[t][v] - r2v[v] - e2v);   // exp(-d)
      Mmat[(size_t)(rbase + v) * KC + col0 + 16 * t] = (__bf16)val;
    }
  }
}

// ---------------------------------------------------------------------------
// Sinkhorn (column-factor-only formulation; global/row scalars dropped since
// they never change a per-row argmax):
//   a[n] = 1 / sum_k M[n][k]*b[k]   ;   b[k] = 1 / sum_n M[n][k]*a[n]
// ---------------------------------------------------------------------------
__global__ void init_b(float* __restrict__ b) {
  b[blockIdx.x * 256 + threadIdx.x] = 1.0f;
}

__global__ __launch_bounds__(256) void rowsum_inv(const __bf16* __restrict__ Mmat,
    const float* __restrict__ bvec, float* __restrict__ avec) {
  int lane = threadIdx.x & 31;
  int n = blockIdx.x * 8 + (threadIdx.x >> 5);
  const __bf16* Mrow = Mmat + (size_t)n * KC;
  float s = 0.f;
  #pragma unroll
  for (int base = 0; base < KC; base += 256) {      // 32 lanes x 8 contiguous
    int k0 = base + lane * 8;
    v8bf m = *(const v8bf*)(Mrow + k0);
    const float* bp = bvec + k0;
    #pragma unroll
    for (int i = 0; i < 8; ++i) s += (float)m[i] * bp[i];
  }
  #pragma unroll
  for (int off = 16; off > 0; off >>= 1) s += __shfl_xor(s, off, 32);
  if (lane == 0) avec[n] = 1.0f / s;
}

__global__ __launch_bounds__(256) void colsum_partial(const __bf16* __restrict__ Mmat,
    const float* __restrict__ avec, float* __restrict__ parts) {
  int col  = blockIdx.x * 512 + threadIdx.x * 2;    // 2 adjacent cols -> 4B loads
  int row0 = blockIdx.y * 128;
  float s0 = 0.f, s1 = 0.f;
  for (int r = 0; r < 128; ++r) {
    const __bf16* p = Mmat + (size_t)(row0 + r) * KC + col;
    __builtin_prefetch(p + 8 * KC, 0, 0);           // global_prefetch ahead of row walk
    v2bf m = *(const v2bf*)p;
    float av = avec[row0 + r];
    s0 += (float)m[0] * av;
    s1 += (float)m[1] * av;
  }
  parts[(size_t)blockIdx.y * KC + col]     = s0;    // deterministic two-phase (no atomics)
  parts[(size_t)blockIdx.y * KC + col + 1] = s1;
}

__global__ __launch_bounds__(256) void colsum_finalize(const float* __restrict__ parts,
    float* __restrict__ bvec) {
  int k = blockIdx.x * 256 + threadIdx.x;
  float s = 0.f;
  for (int g = 0; g < 256; ++g) s += parts[(size_t)g * KC + k];
  bvec[k] = 1.0f / s;
}

// ---------------------------------------------------------------------------
// idx[n] = argmax_k M[n][k]*b[k] (first-max tie-break, matches jnp.argmax);
// then xhat += E[idx], residual -= E[idx]; fused prep of next level's bf16
// residual and row norms.
// ---------------------------------------------------------------------------
__global__ __launch_bounds__(256) void argmax_update(const __bf16* __restrict__ Mmat,
    const float* __restrict__ bvec, const float* __restrict__ Elvl,
    float* __restrict__ resid, float* __restrict__ xhat,
    __bf16* __restrict__ rbf, float* __restrict__ rn2) {
  __shared__ float sv[256];
  __shared__ int   si[256];
  int n = blockIdx.x, tid = threadIdx.x;
  const __bf16* Mrow = Mmat + (size_t)n * KC;
  // Thread t covers the contiguous chunk k = 4t..4t+3 (256*4 == KC); within a
  // chunk ascending-k strict '>' keeps the first max, and the cross-thread
  // lower-index tie-break preserves global first-occurrence semantics.
  int k0 = tid * 4;
  v4bf mv = *(const v4bf*)(Mrow + k0);
  float best = -3.4e38f; int bi = 0x7FFFFFFF;
  #pragma unroll
  for (int i = 0; i < 4; ++i) {
    float v = (float)mv[i] * bvec[k0 + i];
    if (v > best) { best = v; bi = k0 + i; }
  }
  sv[tid] = best; si[tid] = bi; __syncthreads();
  for (int s = 128; s > 0; s >>= 1) {
    if (tid < s) {
      float ov = sv[tid + s]; int oi = si[tid + s];
      if (ov > sv[tid] || (ov == sv[tid] && oi < si[tid])) { sv[tid] = ov; si[tid] = oi; }
    }
    __syncthreads();
  }
  int idx = si[0];
  const float* Erow = Elvl + (size_t)idx * DD;
  float sq = 0.f;
  for (int d = tid; d < DD; d += 256) {
    float e = Erow[d];
    size_t o = (size_t)n * DD + d;
    xhat[o] += e;
    float r = resid[o] - e;
    resid[o] = r;
    rbf[o]   = (__bf16)r;
    sq += r * r;
  }
  __syncthreads();
  sv[tid] = sq; __syncthreads();
  for (int s = 128; s > 0; s >>= 1) {
    if (tid < s) sv[tid] += sv[tid + s];
    __syncthreads();
  }
  if (tid == 0) rn2[n] = sv[0];
}

// ---------------------------------------------------------------------------
extern "C" void kernel_launch(void* const* d_in, const int* in_sizes, int n_in,
                              void* d_out, int out_size, void* d_ws, size_t ws_size,
                              hipStream_t stream) {
  const float* x  = (const float*)d_in[0];        // [N, D]
  const float* cb = (const float*)d_in[1];        // [NQ, K, D]
  // d_in[2] = sigma: no forward effect — straight-through makes the noisy
  // softmax cancel (one_hot + y_soft - stopgrad(y_soft) == one_hot fwd).
  float* xhat = (float*)d_out;                    // [N, D]

  char* w = (char*)d_ws;
  size_t off = 0;
  __bf16* Mmat  = (__bf16*)(w + off); off += (size_t)NN * KC * 2;      //  64 MB
  float*  resid = (float*)(w + off);  off += (size_t)NN * DD * 4;      //  64 MB
  __bf16* rbf   = (__bf16*)(w + off); off += (size_t)NN * DD * 2;      //  32 MB
  __bf16* EbfT  = (__bf16*)(w + off); off += (size_t)NQ * KC * DD * 2; //   4 MB
  float*  e2    = (float*)(w + off);  off += (size_t)NQ * KC * 4;
  float*  rn2   = (float*)(w + off);  off += (size_t)NN * 4;
  float*  avec  = (float*)(w + off);  off += (size_t)NN * 4;
  float*  bvec  = (float*)(w + off);  off += (size_t)KC * 4;
  float*  parts = (float*)(w + off);  off += (size_t)256 * KC * 4;     //   1 MB
  (void)ws_size; (void)in_sizes; (void)n_in; (void)out_size;

  prep_codebooks<<<NQ * KC, 256, 0, stream>>>(cb, EbfT, e2);
  init_x<<<NN, 256, 0, stream>>>(x, resid, xhat, rbf, rn2);

  for (int level = 0; level < NQ; ++level) {
    dist_exp_gemm<<<(NN / 16) * (KC / 64) / 8, 256, 0, stream>>>(
        rbf, EbfT + (size_t)level * DD * KC, rn2, e2 + level * KC, Mmat);
    init_b<<<KC / 256, 256, 0, stream>>>(bvec);
    for (int it = 0; it < 3; ++it) {
      rowsum_inv<<<NN / 8, 256, 0, stream>>>(Mmat, bvec, avec);
      colsum_partial<<<dim3(KC / 512, NN / 128), 256, 0, stream>>>(Mmat, avec, parts);
      colsum_finalize<<<KC / 256, 256, 0, stream>>>(parts, bvec);
    }
    argmax_update<<<NN, 256, 0, stream>>>(
        Mmat, bvec, cb + (size_t)level * KC * DD, resid, xhat, rbf, rn2);
  }
}
